// MyRNN_29111288333263
// MI455X (gfx1250) — matrix operand order repaired and checked by reading, not compiled, not run on hardware
//
#include <hip/hip_runtime.h>
#include <hip/hip_bf16.h>

// ---------------- problem constants (from reference) ----------------
#define BATCH   2048
#define SEQLEN  80
#define EMBED   100
#define EPAD    128           // embed padded to multiple of 32 for K loop
#define UNITS   1024
#define NGATE   4096          // 4*UNITS
#define KL0     (UNITS + EPAD)   // 1152 : [h0 | x]
#define KL1     (UNITS + UNITS)  // 2048 : [h0_new | h1]

// ---------------- tiling ----------------
#define MT 128
#define NT 128
#define KT 32
#define TPB 256               // 8 wave32
#define LDSS 40               // LDS row stride (ushorts) for staged tiles

typedef __attribute__((ext_vector_type(16))) __bf16 v16bf;
typedef __attribute__((ext_vector_type(8)))  float  v8f;
typedef int v4i __attribute__((vector_size(16)));               // matches builtin decl
typedef __attribute__((address_space(1))) v4i* gv4i_p;          // global (as1) v4i*
typedef __attribute__((address_space(3))) v4i* lv4i_p;          // LDS (as3) v4i*

union Frag { v16bf v; uint4 q[2]; };

// ---- CDNA5 async global->LDS path (guarded; sync fallback keeps compile safe)
#if defined(__has_builtin)
#if __has_builtin(__builtin_amdgcn_global_load_async_to_lds_b128)
#define USE_ASYNC 1
#endif
#endif

#ifndef USE_ASYNC
#define USE_ASYNC 0
#endif

#if USE_ASYNC
#if __has_builtin(__builtin_amdgcn_s_wait_asynccnt)
#define ASYNC_WAIT(n) __builtin_amdgcn_s_wait_asynccnt(n)
#else
#define ASYNC_WAIT(n) asm volatile("s_wait_asynccnt %0" ::"n"(n))
#endif
#else
#define ASYNC_WAIT(n) ((void)0)
#endif

__device__ __forceinline__ unsigned short f2bf(float x) {
    unsigned int u = __float_as_uint(x);
    u += 0x7FFFu + ((u >> 16) & 1u);      // round-to-nearest-even
    return (unsigned short)(u >> 16);
}
__device__ __forceinline__ float sigm(float x) { return 1.0f / (1.0f + __expf(-x)); }

// Stage one 128x32 bf16 tile (row-major, compile-time leading dim) into LDS.
// Each of 256 threads moves two 16-byte chunks: rows sr and sr+64.
template<int LDA>
__device__ __forceinline__ void stage128x32(const unsigned short* __restrict__ src,
                                            unsigned short (*dst)[LDSS],
                                            int sr, int scc)
{
#pragma unroll
    for (int h = 0; h < 2; ++h) {
        int r = sr + h * 64;
        const unsigned short* g = src + (size_t)r * LDA + scc;
        unsigned short* l = &dst[r][scc];
#if USE_ASYNC
        __builtin_amdgcn_global_load_async_to_lds_b128(
            (gv4i_p)(unsigned long long)g,
            (lv4i_p)(unsigned int)(unsigned long long)l, 0, 0);
#else
        *(uint4*)l = *(const uint4*)g;
#endif
    }
}

// ============================================================================
// Fused LSTM layer step:  Z = [Aseg0 | Aseg1] @ Bt^T + bias ; gates ; c,h update
//   Aseg0 : [BATCH][K0] bf16 row-major   (h0_prev  or  h0_new)
//   Aseg1 : [BATCH][K1] bf16 row-major   (x_t  or  h1_prev)
//   Bt    : [NGATE][K0+K1] bf16, gate-interleaved columns  n' = 4*u + g
// ============================================================================
template<int K0, int K1>
__global__ __launch_bounds__(TPB)
void lstm_step_kernel(const unsigned short* __restrict__ Aseg0,
                      const unsigned short* __restrict__ Aseg1,
                      const unsigned short* __restrict__ Bt,
                      const float* __restrict__ biasr,
                      float* __restrict__ c,
                      unsigned short* __restrict__ h_out,
                      float* __restrict__ h_out_f32)
{
    constexpr int KTOT = K0 + K1;
    constexpr int NK   = KTOT / KT;
    static_assert((NK & 1) == 0, "K loop unrolled by 2");

    __shared__ __align__(16) unsigned char lds_raw[MT * 132 * 4];  // 67.5 KB
    // double-buffered staging, aliased below the Z tile
    unsigned short (*As0)[LDSS] = (unsigned short(*)[LDSS])(lds_raw);
    unsigned short (*As1)[LDSS] = (unsigned short(*)[LDSS])(lds_raw + 1 * MT * LDSS * 2);
    unsigned short (*Bs0)[LDSS] = (unsigned short(*)[LDSS])(lds_raw + 2 * MT * LDSS * 2);
    unsigned short (*Bs1)[LDSS] = (unsigned short(*)[LDSS])(lds_raw + 3 * MT * LDSS * 2);
    float (*zs)[132] = (float(*)[132])lds_raw;                     // reused after K loop

    const int tid   = threadIdx.x;
    const int lane  = tid & 31;
    const int wave  = tid >> 5;
    const int wmrow = (wave & 1) * 64;   // 2 waves down M
    const int wncol = (wave >> 1) * 32;  // 4 waves across N
    const int r16   = lane & 15;
    const int ka    = (lane >> 4) * 8;   // A frag K base (elements)
    const int kb    = (lane >> 4) * 16;  // unused: B frags also load at ka
    const int sr    = tid >> 2;          // staging row (0..63)
    const int scc   = (tid & 3) * 8;     // staging col chunk (elements)

    const int mbase = blockIdx.y * MT;   // batch rows
    const int n0    = blockIdx.x * NT;   // interleaved gate columns

    const unsigned short* A0p = Aseg0 + (size_t)mbase * K0;
    const unsigned short* A1p = Aseg1 + (size_t)mbase * K1;
    const unsigned short* Bp  = Bt + (size_t)n0 * KTOT;

    v8f acc[4][2] = {};

    // issue async stage of chunk k into the given (compile-time) buffers
    auto stage_chunk = [&](int k, unsigned short (*Ad)[LDSS],
                           unsigned short (*Bd)[LDSS]) {
        if (k < K0) stage128x32<K0>(A0p + k, Ad, sr, scc);
        else        stage128x32<K1>(A1p + (k - K0), Ad, sr, scc);
        stage128x32<KTOT>(Bp + k, Bd, sr, scc);
    };

    // 8 WMMAs on one staged chunk (compile-time buffers)
    auto compute_chunk = [&](unsigned short (*Ac)[LDSS],
                             unsigned short (*Bc)[LDSS]) {
        Frag a[4], b[2];
#pragma unroll
        for (int i = 0; i < 4; ++i) {
            int row = wmrow + i * 16 + r16;
            a[i].q[0] = *(const uint4*)&Ac[row][ka];
            a[i].q[1] = *(const uint4*)&Ac[row][ka + 16];
        }
#pragma unroll
        for (int j = 0; j < 2; ++j) {
            int col = wncol + j * 16 + r16;
            b[j].q[0] = *(const uint4*)&Bc[col][ka];
            b[j].q[1] = *(const uint4*)&Bc[col][ka + 16];
        }
#pragma unroll
        for (int i = 0; i < 4; ++i)
#pragma unroll
            for (int j = 0; j < 2; ++j)
                acc[i][j] = __builtin_amdgcn_wmma_f32_16x16x32_bf16(
                    false, a[i].v, false, b[j].v, (short)0, acc[i][j], false, false);
    };

    // ---- software-pipelined K loop, unrolled x2 so buffers are static ----
    stage_chunk(0, As0, Bs0);
    for (int ck = 0; ck < NK; ck += 2) {
        stage_chunk(ck + 1, As1, Bs1);   // NK even: ck+1 always valid
        ASYNC_WAIT(4);                   // in-order: chunk ck has landed
        __syncthreads();
        compute_chunk(As0, Bs0);
        __syncthreads();

        if (ck + 2 < NK) {
            stage_chunk(ck + 2, As0, Bs0);
            ASYNC_WAIT(4);               // chunk ck+1 has landed
        } else {
            ASYNC_WAIT(0);
        }
        __syncthreads();
        compute_chunk(As1, Bs1);
        __syncthreads();
    }

    // ---- spill Z tile to LDS (C/D layout: vgpr v -> M = v + 8*(lane>=16)) ----
    const int mhalf = (lane >> 4) * 8;
#pragma unroll
    for (int i = 0; i < 4; ++i)
#pragma unroll
        for (int j = 0; j < 2; ++j)
#pragma unroll
            for (int v = 0; v < 8; ++v)
                zs[wmrow + i * 16 + v + mhalf][wncol + j * 16 + r16] = acc[i][j][v];
    __syncthreads();

    // ---- LSTM pointwise: columns 4u..4u+3 are gates i,f,g,o of unit u ----
    for (int idx = tid; idx < MT * (NT / 4); idx += TPB) {
        int r  = idx >> 5;          // row in tile
        int ul = idx & 31;          // unit in tile
        int cb = ul * 4;
        float zi = zs[r][cb + 0] + biasr[n0 + cb + 0];
        float zf = zs[r][cb + 1] + biasr[n0 + cb + 1];
        float zg = zs[r][cb + 2] + biasr[n0 + cb + 2];
        float zo = zs[r][cb + 3] + biasr[n0 + cb + 3];
        int bg = mbase + r;
        int ug = (n0 >> 2) + ul;
        size_t ci = (size_t)bg * UNITS + ug;
        float cv = c[ci];
        float gi = sigm(zi), gf = sigm(zf), gg = tanhf(zg), go = sigm(zo);
        float cn = gf * cv + gi * gg;
        c[ci] = cn;
        float hn = go * tanhf(cn);
        h_out[ci] = f2bf(hn);
        if (h_out_f32) h_out_f32[ci] = hn;
    }
}

// ============================================================================
// Prep kernels (run once per launch; weights -> bf16, N-major, gate-interleaved)
// ============================================================================
__global__ void prep_bt_l0(const float* __restrict__ W0, const float* __restrict__ U0,
                           unsigned short* __restrict__ Bt0)
{
    size_t idx = (size_t)blockIdx.x * TPB + threadIdx.x;
    if (idx >= (size_t)NGATE * KL0) return;
    int k  = (int)(idx % KL0);
    int np = (int)(idx / KL0);
    int u = np >> 2, g = np & 3;
    int col = g * UNITS + u;
    float v;
    if (k < UNITS) v = U0[(size_t)k * NGATE + col];
    else {
        int kw = k - UNITS;
        v = (kw < EMBED) ? W0[(size_t)kw * NGATE + col] : 0.0f;
    }
    Bt0[idx] = f2bf(v);
}

__global__ void prep_bt_l1(const float* __restrict__ W1, const float* __restrict__ U1,
                           unsigned short* __restrict__ Bt1)
{
    size_t idx = (size_t)blockIdx.x * TPB + threadIdx.x;
    if (idx >= (size_t)NGATE * KL1) return;
    int k  = (int)(idx % KL1);
    int np = (int)(idx / KL1);
    int u = np >> 2, g = np & 3;
    int col = g * UNITS + u;
    float v = (k < UNITS) ? W1[(size_t)k * NGATE + col]
                          : U1[(size_t)(k - UNITS) * NGATE + col];
    Bt1[idx] = f2bf(v);
}

__global__ void prep_bias(const float* __restrict__ b0, const float* __restrict__ b1,
                          float* __restrict__ b0r, float* __restrict__ b1r)
{
    int np = blockIdx.x * TPB + threadIdx.x;
    if (np >= NGATE) return;
    int u = np >> 2, g = np & 3;
    b0r[np] = b0[g * UNITS + u];
    b1r[np] = b1[g * UNITS + u];
}

// Xemb[t][b][e] = bf16(emb[inputs[b][t]][e]), zero-padded e in [100,128)
__global__ void embed_gather(const int* __restrict__ inputs, const float* __restrict__ emb,
                             unsigned short* __restrict__ X)
{
    size_t idx = (size_t)blockIdx.x * TPB + threadIdx.x;
    if (idx >= (size_t)SEQLEN * BATCH * EPAD) return;
    int e = (int)(idx & (EPAD - 1));
    size_t tb = idx >> 7;
    int b = (int)(tb & (BATCH - 1));
    int t = (int)(tb >> 11);
    int w = inputs[(size_t)b * SEQLEN + t];
    X[idx] = (e < EMBED) ? f2bf(emb[(size_t)w * EMBED + e]) : (unsigned short)0;
}

// out[b] = sigmoid(h1[b,:] . fcW + fcb)
__global__ __launch_bounds__(TPB)
void final_fc(const float* __restrict__ h1, const float* __restrict__ fcW,
              const float* __restrict__ fcb, float* __restrict__ out)
{
    __shared__ float red[TPB];
    int b = blockIdx.x, tid = threadIdx.x;
    float s = 0.0f;
    for (int k = tid; k < UNITS; k += TPB) s += h1[(size_t)b * UNITS + k] * fcW[k];
    red[tid] = s;
    __syncthreads();
    for (int off = TPB / 2; off > 0; off >>= 1) {
        if (tid < off) red[tid] += red[tid + off];
        __syncthreads();
    }
    if (tid == 0) out[b] = sigm(red[0] + fcb[0]);
}

// ============================================================================
extern "C" void kernel_launch(void* const* d_in, const int* in_sizes, int n_in,
                              void* d_out, int out_size, void* d_ws, size_t ws_size,
                              hipStream_t stream)
{
    (void)in_sizes; (void)n_in; (void)out_size; (void)ws_size;
    const int*   inputs = (const int*)  d_in[0];
    const float* emb    = (const float*)d_in[1];
    const float* W0     = (const float*)d_in[2];
    const float* U0     = (const float*)d_in[3];
    const float* b0     = (const float*)d_in[4];
    const float* W1     = (const float*)d_in[5];
    const float* U1     = (const float*)d_in[6];
    const float* b1     = (const float*)d_in[7];
    const float* fcW    = (const float*)d_in[8];
    const float* fcb    = (const float*)d_in[9];
    float* out = (float*)d_out;

    // ---- workspace carve-up ----
    char* ws = (char*)d_ws;
    size_t off = 0;
    auto carve = [&](size_t bytes) -> void* {
        void* p = ws + off;
        off = (off + bytes + 255) & ~(size_t)255;
        return p;
    };
    unsigned short* Bt0  = (unsigned short*)carve((size_t)NGATE * KL0 * 2);
    unsigned short* Bt1  = (unsigned short*)carve((size_t)NGATE * KL1 * 2);
    float*          b0r  = (float*)carve((size_t)NGATE * 4);
    float*          b1r  = (float*)carve((size_t)NGATE * 4);
    unsigned short* Xemb = (unsigned short*)carve((size_t)SEQLEN * BATCH * EPAD * 2);
    unsigned short* h0p[2], *h1p[2];
    h0p[0] = (unsigned short*)carve((size_t)BATCH * UNITS * 2);
    h0p[1] = (unsigned short*)carve((size_t)BATCH * UNITS * 2);
    h1p[0] = (unsigned short*)carve((size_t)BATCH * UNITS * 2);
    h1p[1] = (unsigned short*)carve((size_t)BATCH * UNITS * 2);
    float* c0    = (float*)carve((size_t)BATCH * UNITS * 4);
    float* c1    = (float*)carve((size_t)BATCH * UNITS * 4);
    float* h1f32 = (float*)carve((size_t)BATCH * UNITS * 4);

    // ---- zero initial state (graph-capture-safe memset nodes) ----
    (void)hipMemsetAsync(h0p[0], 0, (size_t)BATCH * UNITS * 2, stream);
    (void)hipMemsetAsync(h1p[0], 0, (size_t)BATCH * UNITS * 2, stream);
    (void)hipMemsetAsync(c0,     0, (size_t)BATCH * UNITS * 4, stream);
    (void)hipMemsetAsync(c1,     0, (size_t)BATCH * UNITS * 4, stream);

    // ---- one-time prep ----
    {
        size_t n = (size_t)NGATE * KL0;
        prep_bt_l0<<<(unsigned)((n + TPB - 1) / TPB), TPB, 0, stream>>>(W0, U0, Bt0);
    }
    {
        size_t n = (size_t)NGATE * KL1;
        prep_bt_l1<<<(unsigned)((n + TPB - 1) / TPB), TPB, 0, stream>>>(W1, U1, Bt1);
    }
    prep_bias<<<(NGATE + TPB - 1) / TPB, TPB, 0, stream>>>(b0, b1, b0r, b1r);
    {
        size_t n = (size_t)SEQLEN * BATCH * EPAD;
        embed_gather<<<(unsigned)((n + TPB - 1) / TPB), TPB, 0, stream>>>(inputs, emb, Xemb);
    }

    // ---- recurrence: 80 steps x 2 layers, ping-pong h buffers ----
    dim3 grid(NGATE / NT, BATCH / MT);   // 32 x 16 = 512 workgroups
    for (int t = 0; t < SEQLEN; ++t) {
        int rd = t & 1, wr = (t + 1) & 1;
        const unsigned short* xt = Xemb + (size_t)t * BATCH * EPAD;
        // layer 0: A = [h0_prev | x_t]
        lstm_step_kernel<UNITS, EPAD><<<grid, TPB, 0, stream>>>(
            h0p[rd], xt, Bt0, b0r, c0, h0p[wr], nullptr);
        // layer 1: A = [h0_new | h1_prev]
        lstm_step_kernel<UNITS, UNITS><<<grid, TPB, 0, stream>>>(
            h0p[wr], h1p[rd], Bt1, b1r, c1, h1p[wr],
            (t == SEQLEN - 1) ? h1f32 : nullptr);
    }

    final_fc<<<BATCH, TPB, 0, stream>>>(h1f32, fcW, fcb, out);
}
